// MoEContradictionClassifier_42829413876264
// MI455X (gfx1250) — matrix operand order repaired and checked by reading, not compiled, not run on hardware
//
#include <hip/hip_runtime.h>
#include <math.h>

// ---------------- problem constants ----------------
#define VV     30522
#define DD     768
#define HH     12
#define DH     64
#define FF     3072
#define LLAY   2
#define EE     4
#define BBATCH 64
#define SS     128
#define NT     (BBATCH * SS)   // 8192 tokens
#define GHID   256
#define NOUT   3

#define ACT_NONE 0
#define ACT_GELU 1
#define ACT_RELU 2

typedef __bf16 bf16_t;
typedef __attribute__((ext_vector_type(16))) __bf16 v16bf;
typedef __attribute__((ext_vector_type(8)))  __bf16 v8bf;
typedef __attribute__((ext_vector_type(4)))  __bf16 v4bf;
typedef __attribute__((ext_vector_type(2)))  __bf16 v2bf;
typedef __attribute__((ext_vector_type(8)))  float  v8f;
typedef __attribute__((ext_vector_type(4)))  float  v4f;

// ---------------- device helpers ----------------
__device__ __forceinline__ float block_sum(float v, float* sh, int nthreads) {
    int t = threadIdx.x;
    sh[t] = v; __syncthreads();
    for (int o = nthreads >> 1; o > 0; o >>= 1) {
        if (t < o) sh[t] += sh[t + o];
        __syncthreads();
    }
    float r = sh[0]; __syncthreads();
    return r;
}

__device__ __forceinline__ float gelu_tanh(float x) {
    float u = 0.7978845608028654f * (x + 0.044715f * x * x * x);
    return 0.5f * x * (1.0f + tanhf(u));
}

// ---------------- embedding + post-LN ----------------
__global__ void embed_ln_kernel(const int* __restrict__ ids,
                                const float* __restrict__ emb,
                                const float* __restrict__ pos,
                                const float* __restrict__ g,
                                const float* __restrict__ bb,
                                float* __restrict__ X) {
    __shared__ float sh[256];
    int t = blockIdx.x;
    int s = t % SS;
    int id = ids[t];
    const float* er = emb + (size_t)id * DD;
    const float* pr = pos + (size_t)s * DD;
    float v[3]; float sum = 0.f;
#pragma unroll
    for (int i = 0; i < 3; ++i) {
        int d = i * 256 + threadIdx.x;
        v[i] = er[d] + pr[d];
        sum += v[i];
    }
    float mean = block_sum(sum, sh, 256) * (1.0f / DD);
    float vs = 0.f;
#pragma unroll
    for (int i = 0; i < 3; ++i) { float c = v[i] - mean; vs += c * c; }
    float var = block_sum(vs, sh, 256) * (1.0f / DD);
    float rstd = rsqrtf(var + 1e-5f);
    float* xr = X + (size_t)t * DD;
#pragma unroll
    for (int i = 0; i < 3; ++i) {
        int d = i * 256 + threadIdx.x;
        xr[d] = (v[i] - mean) * rstd * g[d] + bb[d];
    }
}

// ---------------- residual add + LN ----------------
__global__ void add_ln_kernel(float* __restrict__ X,
                              const float* __restrict__ P,
                              const float* __restrict__ g,
                              const float* __restrict__ bb) {
    __shared__ float sh[256];
    int t = blockIdx.x;
    float* xr = X + (size_t)t * DD;
    const float* pr = P + (size_t)t * DD;
    float v[3]; float sum = 0.f;
#pragma unroll
    for (int i = 0; i < 3; ++i) {
        int d = i * 256 + threadIdx.x;
        v[i] = xr[d] + pr[d];
        sum += v[i];
    }
    float mean = block_sum(sum, sh, 256) * (1.0f / DD);
    float vs = 0.f;
#pragma unroll
    for (int i = 0; i < 3; ++i) { float c = v[i] - mean; vs += c * c; }
    float var = block_sum(vs, sh, 256) * (1.0f / DD);
    float rstd = rsqrtf(var + 1e-5f);
#pragma unroll
    for (int i = 0; i < 3; ++i) {
        int d = i * 256 + threadIdx.x;
        xr[d] = (v[i] - mean) * rstd * g[d] + bb[d];
    }
}

// ==================================================================
// LDS-staged, double-buffered WMMA bf16 GEMM
//   C[M,N] = act(A[M,K] @ W[K,N] + bias[N])
// block = 256 threads = 8 waves (2 along M x 4 along N)
// block tile = 64(M) x 128(N); wave tile = 32x32 (4 wmma accs); K step = 32
// grid: (N/128, M/64). M % 64 == 0, N % 128 == 0, K % 32 == 0.
// ==================================================================
#define MT   64
#define NTl  128
#define KT   32

// Staging: issue ALL global loads first (independent, overlap in flight),
// then convert f32->bf16 and store to LDS. B tile is loaded as k-pairs so
// each (k,k+1,n) pair packs into one 4-byte aligned ds_store_b32.
__device__ __forceinline__ void stage_tiles(const float* __restrict__ A,
                                            const float* __restrict__ W,
                                            int K, int N, int tileM, int tileN, int kb,
                                            __bf16 (*Al)[KT], __bf16 (*Bl)[KT]) {
    int t = threadIdx.x;

    // ---- phase 1: issue all global loads ----
    // A tile: 64 rows x 32 k -> 512 float4 units, 2 per thread (coalesced)
    v4f av[2];
    int arow[2], aseg[2];
#pragma unroll
    for (int u = 0; u < 2; ++u) {
        int idx = t + u * 256;            // 0..511
        arow[u] = idx >> 3;               // 0..63
        aseg[u] = idx & 7;                // k = seg*4
        av[u] = *(const v4f*)(A + (size_t)(tileM + arow[u]) * K + kb + aseg[u] * 4);
    }
    // B tile: 32 k x 128 n, loaded as 16 k-pairs x 32 n-segments = 512 units, 2/thread
    v4f bv0[2], bv1[2];
    int bkp[2], bn0[2];
#pragma unroll
    for (int u = 0; u < 2; ++u) {
        int idx = t + u * 256;            // 0..511
        bkp[u] = idx >> 5;                // 0..15 (k-pair)
        bn0[u] = (idx & 31) * 4;          // n base
        const float* wr = W + (size_t)(kb + 2 * bkp[u]) * N + tileN + bn0[u];
        bv0[u] = __builtin_nontemporal_load((const v4f*)wr);        // row k
        bv1[u] = __builtin_nontemporal_load((const v4f*)(wr + N));  // row k+1
    }

    // ---- phase 2: convert + LDS stores ----
#pragma unroll
    for (int u = 0; u < 2; ++u) {
        v4bf p;
        p[0] = (bf16_t)av[u][0]; p[1] = (bf16_t)av[u][1];
        p[2] = (bf16_t)av[u][2]; p[3] = (bf16_t)av[u][3];
        *(v4bf*)&Al[arow[u]][aseg[u] * 4] = p;   // 8-byte LDS store
    }
#pragma unroll
    for (int u = 0; u < 2; ++u) {
#pragma unroll
        for (int i = 0; i < 4; ++i) {
            v2bf pr;
            pr[0] = (bf16_t)bv0[u][i];           // k   = 2*bkp
            pr[1] = (bf16_t)bv1[u][i];           // k+1 = 2*bkp+1
            *(v2bf*)&Bl[bn0[u] + i][2 * bkp[u]] = pr;   // 4-byte LDS store
        }
    }
}

__global__ __launch_bounds__(256) void gemm_bf16_wmma_kernel(
        const float* __restrict__ A, const float* __restrict__ W,
        const float* __restrict__ bias, float* __restrict__ C,
        int M, int N, int K, int act) {
    __shared__ __bf16 Alds[2][MT][KT];    // 2 x 4 KB
    __shared__ __bf16 Blds[2][NTl][KT];   // 2 x 8 KB (transposed: [n][k])

    int lane = threadIdx.x & 31;
    int wid  = threadIdx.x >> 5;          // 0..7
    int wm   = wid >> 2;                  // 0..1 -> rows wm*32
    int wn   = wid & 3;                   // 0..3 -> cols wn*32
    int tileM = blockIdx.y * MT;
    int tileN = blockIdx.x * NTl;

    int rsel = lane >> 4;                 // lane half select
    int lrow = lane & 15;

    v8f acc[4] = {{}, {}, {}, {}};        // [fm*2+fn]

    stage_tiles(A, W, K, N, tileM, tileN, 0, Alds[0], Blds[0]);

    int buf = 0;
    for (int kb = 0; kb < K; kb += KT, buf ^= 1) {
        __syncthreads();                  // staged tile [buf] visible
        if (kb + KT < K)
            stage_tiles(A, W, K, N, tileM, tileN, kb + KT, Alds[buf ^ 1], Blds[buf ^ 1]);

        // ---- build fragments from LDS (16B ds reads) ----
        v16bf afr[2], bfr[2];
#pragma unroll
        for (int fm = 0; fm < 2; ++fm) {
            int row = wm * 32 + fm * 16 + lrow;
            v8bf lo = *(const v8bf*)&Alds[buf][row][rsel * 8];        // k = rsel*8 .. +7
            v8bf hi = *(const v8bf*)&Alds[buf][row][16 + rsel * 8];   // k = 16+rsel*8 .. +7
            afr[fm] = __builtin_shufflevector(lo, hi, 0, 1, 2, 3, 4, 5, 6, 7,
                                              8, 9, 10, 11, 12, 13, 14, 15);
        }
#pragma unroll
        for (int fn = 0; fn < 2; ++fn) {
            int col = wn * 32 + fn * 16 + lrow;
            v8bf lo = *(const v8bf*)&Blds[buf][col][rsel * 8];        // k = rsel*8 .. +7
            v8bf hi = *(const v8bf*)&Blds[buf][col][16 + rsel * 8];   // k = 16+rsel*8 .. +7
            bfr[fn] = __builtin_shufflevector(lo, hi, 0, 1, 2, 3, 4, 5, 6, 7,
                                              8, 9, 10, 11, 12, 13, 14, 15);
        }
        // ---- 4 WMMAs, A/B fragment reuse ----
#pragma unroll
        for (int fm = 0; fm < 2; ++fm)
#pragma unroll
            for (int fn = 0; fn < 2; ++fn)
                acc[fm * 2 + fn] = __builtin_amdgcn_wmma_f32_16x16x32_bf16(
                    false, afr[fm], false, bfr[fn],
                    (short)0, acc[fm * 2 + fn], false, false);
    }

    // ---- epilogue: bias + activation, D-layout stores ----
#pragma unroll
    for (int fm = 0; fm < 2; ++fm) {
#pragma unroll
        for (int fn = 0; fn < 2; ++fn) {
            int col = tileN + wn * 32 + fn * 16 + lrow;
            float bv = bias ? bias[col] : 0.0f;
            v8f a = acc[fm * 2 + fn];
#pragma unroll
            for (int j = 0; j < 8; ++j) {
                int r = tileM + wm * 32 + fm * 16 + (rsel ? j + 8 : j);
                float val = a[j] + bv;
                if (act == ACT_GELU)      val = gelu_tanh(val);
                else if (act == ACT_RELU) val = fmaxf(val, 0.0f);
                C[(size_t)r * N + col] = val;
            }
        }
    }
}

// ---------------- attention: fused scores + mask + softmax ----------------
__global__ void attn_softmax_kernel(const float* __restrict__ Q,
                                    const float* __restrict__ Kb,
                                    const int* __restrict__ mask,
                                    float* __restrict__ ATT) {
    __shared__ float sh[128];
    int b = blockIdx.z, h = blockIdx.y, qi = blockIdx.x, ki = threadIdx.x;
    const float* qr = Q + (size_t)(b * SS + qi) * DD + h * DH;
    const float* kr = Kb + (size_t)(b * SS + ki) * DD + h * DH;
    float s = 0.f;
#pragma unroll 8
    for (int c = 0; c < DH; ++c) s += qr[c] * kr[c];
    s = s * 0.125f + (1.0f - (float)mask[b * SS + ki]) * (-1e9f);
    sh[ki] = s; __syncthreads();
    for (int o = 64; o > 0; o >>= 1) { if (ki < o) sh[ki] = fmaxf(sh[ki], sh[ki + o]); __syncthreads(); }
    float mx = sh[0]; __syncthreads();
    float e = expf(s - mx);
    sh[ki] = e; __syncthreads();
    for (int o = 64; o > 0; o >>= 1) { if (ki < o) sh[ki] += sh[ki + o]; __syncthreads(); }
    float inv = 1.0f / sh[0];
    ATT[(((size_t)b * HH + h) * SS + qi) * SS + ki] = e * inv;
}

// ---------------- ctx = attn @ V ----------------
__global__ void ctx_kernel(const float* __restrict__ ATT,
                           const float* __restrict__ Vb,
                           float* __restrict__ CTX) {
    int b = blockIdx.z, h = blockIdx.y, qi = blockIdx.x, c = threadIdx.x;
    const float* ar = ATT + (((size_t)b * HH + h) * SS + qi) * SS;
    float acc = 0.f;
#pragma unroll 8
    for (int ki = 0; ki < SS; ++ki)
        acc += ar[ki] * Vb[(size_t)(b * SS + ki) * DD + h * DH + c];
    CTX[(size_t)(b * SS + qi) * DD + h * DH + c] = acc;
}

// ---------------- extract CLS ----------------
__global__ void extract_cls_kernel(const float* __restrict__ X, float* __restrict__ cls) {
    int idx = blockIdx.x * 256 + threadIdx.x;
    if (idx >= BBATCH * DD) return;
    int b = idx / DD, d = idx % DD;
    cls[idx] = X[(size_t)(b * SS) * DD + d];
}

// ---------------- generic row LN + activation (n <= 512), in place ----------------
__global__ void ln_act_kernel(float* __restrict__ X, const float* __restrict__ g,
                              const float* __restrict__ bb, int n, int act) {
    __shared__ float sh[256];
    int row = blockIdx.x;
    float* xr = X + (size_t)row * n;
    int d0 = threadIdx.x, d1 = threadIdx.x + 256;
    float v0 = (d0 < n) ? xr[d0] : 0.f;
    float v1 = (d1 < n) ? xr[d1] : 0.f;
    float mean = block_sum(v0 + v1, sh, 256) / (float)n;
    float c0 = (d0 < n) ? v0 - mean : 0.f;
    float c1 = (d1 < n) ? v1 - mean : 0.f;
    float var = block_sum(c0 * c0 + c1 * c1, sh, 256) / (float)n;
    float rstd = rsqrtf(var + 1e-5f);
    if (d0 < n) {
        float y = c0 * rstd * g[d0] + bb[d0];
        if (act == ACT_GELU) y = gelu_tanh(y); else if (act == ACT_RELU) y = fmaxf(y, 0.f);
        xr[d0] = y;
    }
    if (d1 < n) {
        float y = c1 * rstd * g[d1] + bb[d1];
        if (act == ACT_GELU) y = gelu_tanh(y); else if (act == ACT_RELU) y = fmaxf(y, 0.f);
        xr[d1] = y;
    }
}

// ---------------- gate head: logits -> softmax -> top-2 dense weights ----------------
__global__ void gate_head_kernel(const float* __restrict__ Hd,
                                 const float* __restrict__ W1,
                                 const float* __restrict__ b1,
                                 float* __restrict__ probs_out,
                                 float* __restrict__ wout) {
    __shared__ float lg[EE];
    int b = blockIdx.x, e = threadIdx.x;
    if (e < EE) {
        float acc = b1[e];
        const float* hr = Hd + (size_t)b * GHID;
        for (int c = 0; c < GHID; ++c) acc += hr[c] * W1[c * EE + e];
        lg[e] = acc;
    }
    __syncthreads();
    if (threadIdx.x == 0) {
        float mx = lg[0];
        for (int i = 1; i < EE; ++i) mx = fmaxf(mx, lg[i]);
        float ex[EE], sum = 0.f;
        for (int i = 0; i < EE; ++i) { ex[i] = expf(lg[i] - mx); sum += ex[i]; }
        float inv = 1.0f / sum;
        float p[EE];
        int i0 = 0, i1 = -1;
        for (int i = 0; i < EE; ++i) { p[i] = ex[i] * inv; probs_out[b * EE + i] = p[i]; }
        for (int i = 1; i < EE; ++i) if (p[i] > p[i0]) i0 = i;
        for (int i = 0; i < EE; ++i) { if (i == i0) continue; if (i1 < 0 || p[i] > p[i1]) i1 = i; }
        for (int i = 0; i < EE; ++i) wout[b * EE + i] = (i == i0 || i == i1) ? p[i] : 0.f;
    }
}

// ---------------- combined[b,d] = sum_e w[b,e] * ecls[e,b,d] ----------------
__global__ void combine_kernel(const float* __restrict__ w,
                               const float* __restrict__ ecls,
                               float* __restrict__ comb) {
    int idx = blockIdx.x * 256 + threadIdx.x;
    if (idx >= BBATCH * DD) return;
    int b = idx / DD, d = idx % DD;
    float acc = 0.f;
#pragma unroll
    for (int e = 0; e < EE; ++e)
        acc += w[b * EE + e] * ecls[((size_t)e * BBATCH + b) * DD + d];
    comb[idx] = acc;
}

// ---------------- final logits: (B,256) @ (256,3) + b ----------------
__global__ void final_logits_kernel(const float* __restrict__ X,
                                    const float* __restrict__ W2,
                                    const float* __restrict__ b2,
                                    float* __restrict__ out) {
    int b = blockIdx.x, o = threadIdx.x;
    if (o < NOUT) {
        float acc = b2[o];
        const float* xr = X + (size_t)b * 256;
        for (int c = 0; c < 256; ++c) acc += xr[c] * W2[c * NOUT + o];
        out[b * NOUT + o] = acc;
    }
}

// =================== host-side orchestration ===================

// Sorted-key leaf order of one encoder pytree (JAX dict flatten sorts keys):
// 0:W1(L,D,F) 1:W2(L,F,D) 2:Wk 3:Wo 4:Wq 5:Wv (L,D,D)
// 6:b1(L,F) 7:b2 8:bk 9:bo 10:bq 11:bv (L,D)
// 12:emb(V,D) 13:ln1_b 14:ln1_s 15:ln2_b 16:ln2_s (L,D)
// 17:ln_e_b 18:ln_e_s (D) 19:pos(S,D)
static const size_t kEncLeafElems[20] = {
    (size_t)LLAY * DD * FF, (size_t)LLAY * FF * DD,
    (size_t)LLAY * DD * DD, (size_t)LLAY * DD * DD, (size_t)LLAY * DD * DD, (size_t)LLAY * DD * DD,
    (size_t)LLAY * FF, (size_t)LLAY * DD, (size_t)LLAY * DD, (size_t)LLAY * DD,
    (size_t)LLAY * DD, (size_t)LLAY * DD,
    (size_t)VV * DD,
    (size_t)LLAY * DD, (size_t)LLAY * DD, (size_t)LLAY * DD, (size_t)LLAY * DD,
    (size_t)DD, (size_t)DD, (size_t)SS * DD
};

struct EncP { const float* p[20]; };

struct Ws {
    float *X, *QB, *KB, *VB, *CB, *PB, *FB, *AT;
    float *GCLS, *ECLS, *GH, *WCOMB, *COMB, *C0, *C1;
};

static void run_encoder(hipStream_t stream, const EncP& ep,
                        const int* ids, const int* mask, const Ws& ws, float* cls_out) {
    const float* Wff1 = ep.p[0];  const float* Wff2 = ep.p[1];
    const float* Wk = ep.p[2];    const float* Wo = ep.p[3];
    const float* Wq = ep.p[4];    const float* Wv = ep.p[5];
    const float* b1 = ep.p[6];    const float* b2 = ep.p[7];
    const float* bk = ep.p[8];    const float* bo = ep.p[9];
    const float* bq = ep.p[10];   const float* bv = ep.p[11];
    const float* emb = ep.p[12];
    const float* ln1b = ep.p[13]; const float* ln1s = ep.p[14];
    const float* ln2b = ep.p[15]; const float* ln2s = ep.p[16];
    const float* lneb = ep.p[17]; const float* lnes = ep.p[18];
    const float* pos = ep.p[19];

    embed_ln_kernel<<<NT, 256, 0, stream>>>(ids, emb, pos, lnes, lneb, ws.X);

    dim3 gDD(DD / NTl, NT / MT);   // (6,128)
    dim3 gDF(FF / NTl, NT / MT);   // (24,128)
    dim3 attg(SS, HH, BBATCH);

    for (int l = 0; l < LLAY; ++l) {
        size_t wOff = (size_t)l * DD * DD, bOff = (size_t)l * DD;
        gemm_bf16_wmma_kernel<<<gDD, 256, 0, stream>>>(ws.X, Wq + wOff, bq + bOff, ws.QB, NT, DD, DD, ACT_NONE);
        gemm_bf16_wmma_kernel<<<gDD, 256, 0, stream>>>(ws.X, Wk + wOff, bk + bOff, ws.KB, NT, DD, DD, ACT_NONE);
        gemm_bf16_wmma_kernel<<<gDD, 256, 0, stream>>>(ws.X, Wv + wOff, bv + bOff, ws.VB, NT, DD, DD, ACT_NONE);
        attn_softmax_kernel<<<attg, 128, 0, stream>>>(ws.QB, ws.KB, mask, ws.AT);
        ctx_kernel<<<attg, 64, 0, stream>>>(ws.AT, ws.VB, ws.CB);
        gemm_bf16_wmma_kernel<<<gDD, 256, 0, stream>>>(ws.CB, Wo + wOff, bo + bOff, ws.PB, NT, DD, DD, ACT_NONE);
        add_ln_kernel<<<NT, 256, 0, stream>>>(ws.X, ws.PB, ln1s + bOff, ln1b + bOff);
        gemm_bf16_wmma_kernel<<<gDF, 256, 0, stream>>>(ws.X, Wff1 + (size_t)l * DD * FF, b1 + (size_t)l * FF,
                                                       ws.FB, NT, FF, DD, ACT_GELU);
        gemm_bf16_wmma_kernel<<<gDD, 256, 0, stream>>>(ws.FB, Wff2 + (size_t)l * FF * DD, b2 + bOff,
                                                       ws.PB, NT, DD, FF, ACT_NONE);
        add_ln_kernel<<<NT, 256, 0, stream>>>(ws.X, ws.PB, ln2s + bOff, ln2b + bOff);
    }
    extract_cls_kernel<<<(BBATCH * DD + 255) / 256, 256, 0, stream>>>(ws.X, cls_out);
}

extern "C" void kernel_launch(void* const* d_in, const int* in_sizes, int n_in,
                              void* d_out, int out_size, void* d_ws, size_t ws_size,
                              hipStream_t stream) {
    (void)in_sizes; (void)n_in; (void)out_size; (void)ws_size;
    const int* gate_ids  = (const int*)d_in[0];
    const int* gate_mask = (const int*)d_in[1];
    const int* exp_ids   = (const int*)d_in[2];
    const int* exp_mask  = (const int*)d_in[3];

    const int GATING_BASE = 4;    // 20 leaves
    const int EXPERT_BASE = 24;   // 20 leaves, E-stacked
    const int GATE_NET_BASE = 44; // W0, W1, b0, b1, ln_b, ln_s
    const int CLF_BASE = 50;      // W0, W1, W2, b0, b1, b2, ln0_b, ln0_s, ln1_b, ln1_s

    EncP gating;
    for (int i = 0; i < 20; ++i) gating.p[i] = (const float*)d_in[GATING_BASE + i];

    // ---- workspace layout (floats) ----
    float* base = (float*)d_ws;
    Ws ws;
    size_t off = 0;
    ws.X  = base + off; off += (size_t)NT * DD;
    ws.QB = base + off; off += (size_t)NT * DD;
    ws.KB = base + off; off += (size_t)NT * DD;
    ws.VB = base + off; off += (size_t)NT * DD;
    ws.CB = base + off; off += (size_t)NT * DD;
    ws.PB = base + off; off += (size_t)NT * DD;
    ws.FB = base + off; off += (size_t)NT * FF;
    ws.AT = base + off; off += (size_t)BBATCH * HH * SS * SS;
    ws.GCLS  = base + off; off += (size_t)BBATCH * DD;
    ws.ECLS  = base + off; off += (size_t)EE * BBATCH * DD;
    ws.GH    = base + off; off += (size_t)BBATCH * GHID;
    ws.WCOMB = base + off; off += (size_t)BBATCH * EE;
    ws.COMB  = base + off; off += (size_t)BBATCH * DD;
    ws.C0    = base + off; off += (size_t)BBATCH * 512;
    ws.C1    = base + off; off += (size_t)BBATCH * 256;

    float* out_logits = (float*)d_out;                       // (B, 3)
    float* out_probs  = (float*)d_out + BBATCH * NOUT;       // (B, E)

    // ---- gating encoder + gate net ----
    run_encoder(stream, gating, gate_ids, gate_mask, ws, ws.GCLS);

    const float* gW0 = (const float*)d_in[GATE_NET_BASE + 0];
    const float* gW1 = (const float*)d_in[GATE_NET_BASE + 1];
    const float* gb0 = (const float*)d_in[GATE_NET_BASE + 2];
    const float* gb1 = (const float*)d_in[GATE_NET_BASE + 3];
    const float* glnb = (const float*)d_in[GATE_NET_BASE + 4];
    const float* glns = (const float*)d_in[GATE_NET_BASE + 5];

    {
        dim3 g(GHID / NTl, BBATCH / MT);
        gemm_bf16_wmma_kernel<<<g, 256, 0, stream>>>(ws.GCLS, gW0, gb0, ws.GH, BBATCH, GHID, DD, ACT_NONE);
        ln_act_kernel<<<BBATCH, 256, 0, stream>>>(ws.GH, glns, glnb, GHID, ACT_GELU);
        gate_head_kernel<<<BBATCH, 32, 0, stream>>>(ws.GH, gW1, gb1, out_probs, ws.WCOMB);
    }

    // ---- expert encoders (dense; zero-weight experts contribute zero) ----
    for (int e = 0; e < EE; ++e) {
        EncP ep;
        for (int i = 0; i < 20; ++i)
            ep.p[i] = (const float*)d_in[EXPERT_BASE + i] + (size_t)e * kEncLeafElems[i];
        run_encoder(stream, ep, exp_ids, exp_mask, ws, ws.ECLS + (size_t)e * BBATCH * DD);
    }

    // ---- combine + classifier head ----
    combine_kernel<<<(BBATCH * DD + 255) / 256, 256, 0, stream>>>(ws.WCOMB, ws.ECLS, ws.COMB);

    const float* cW0 = (const float*)d_in[CLF_BASE + 0];
    const float* cW1 = (const float*)d_in[CLF_BASE + 1];
    const float* cW2 = (const float*)d_in[CLF_BASE + 2];
    const float* cb0 = (const float*)d_in[CLF_BASE + 3];
    const float* cb1 = (const float*)d_in[CLF_BASE + 4];
    const float* cb2 = (const float*)d_in[CLF_BASE + 5];
    const float* cln0b = (const float*)d_in[CLF_BASE + 6];
    const float* cln0s = (const float*)d_in[CLF_BASE + 7];
    const float* cln1b = (const float*)d_in[CLF_BASE + 8];
    const float* cln1s = (const float*)d_in[CLF_BASE + 9];

    {
        dim3 g0(512 / NTl, BBATCH / MT);
        gemm_bf16_wmma_kernel<<<g0, 256, 0, stream>>>(ws.COMB, cW0, cb0, ws.C0, BBATCH, 512, DD, ACT_NONE);
        ln_act_kernel<<<BBATCH, 256, 0, stream>>>(ws.C0, cln0s, cln0b, 512, ACT_RELU);
        dim3 g1(256 / NTl, BBATCH / MT);
        gemm_bf16_wmma_kernel<<<g1, 256, 0, stream>>>(ws.C0, cW1, cb1, ws.C1, BBATCH, 256, 512, ACT_NONE);
        ln_act_kernel<<<BBATCH, 256, 0, stream>>>(ws.C1, cln1s, cln1b, 256, ACT_RELU);
        final_logits_kernel<<<BBATCH, 32, 0, stream>>>(ws.C1, cW2, cb2, out_logits);
    }
}
